// hbPass_55078660603989
// MI455X (gfx1250) — matrix operand order repaired and checked by reading, not compiled, not run on hardware
//
#include <hip/hip_runtime.h>
#include <hip/hip_bf16.h>

typedef __attribute__((ext_vector_type(16))) _Float16     v16h;
typedef __attribute__((ext_vector_type(8)))  float        v8f;
typedef __attribute__((ext_vector_type(4)))  unsigned int u32x4;

#define HH   130
#define CIN  64
#define COUT 128
#define HWSZ (130*130)

// ---------------- Kernel 1: BatchNorm batch statistics (per channel) -------
__global__ __launch_bounds__(256) void bnstats_kernel(
    const float* __restrict__ x, const float* __restrict__ gamma,
    const float* __restrict__ beta, float* __restrict__ scaleC,
    float* __restrict__ shiftC)
{
    __shared__ float s1[256];
    __shared__ float s2[256];
    const int c = blockIdx.x;           // 0..63
    float sum = 0.f, sum2 = 0.f;
    for (int b = 0; b < 8; ++b) {
        const float* p = x + (size_t)(b * CIN + c) * HWSZ;
        for (int t = threadIdx.x; t < HWSZ; t += 256) {
            float v = p[t];
            sum += v; sum2 += v * v;
        }
    }
    s1[threadIdx.x] = sum; s2[threadIdx.x] = sum2;
    __syncthreads();
    for (int off = 128; off > 0; off >>= 1) {
        if ((int)threadIdx.x < off) {
            s1[threadIdx.x] += s1[threadIdx.x + off];
            s2[threadIdx.x] += s2[threadIdx.x + off];
        }
        __syncthreads();
    }
    if (threadIdx.x == 0) {
        const float inv = 1.f / (8.f * (float)HWSZ);
        float mean = s1[0] * inv;
        float var  = s2[0] * inv - mean * mean;
        float rs   = rsqrtf(var + 1e-4f);
        float sc   = gamma[c] * rs;
        scaleC[c] = sc;
        shiftC[c] = beta[c] - mean * sc;
    }
}

// -------- Kernel 2: weights f32 [co][ci][tap] -> f16 [tap][co][ci] ---------
__global__ __launch_bounds__(256) void wprep_kernel(
    const float* __restrict__ w, _Float16* __restrict__ wh)
{
    int oidx = blockIdx.x * 256 + threadIdx.x;     // over 9*128*64
    if (oidx >= 9 * COUT * CIN) return;
    const int ci  = oidx % CIN;
    const int co  = (oidx / CIN) % COUT;
    const int tap = oidx / (CIN * COUT);
    wh[oidx] = (_Float16)w[((size_t)co * CIN + ci) * 9 + tap];
}

// ------- Kernel 3: fused BN + im2col + bin_active + col2im (per b,y) -------
// img[b][y][x][c] (f16, channel-last) = sign(xbn)*ni(y)*sum_j T[c][j][(x-j)/16]
__global__ __launch_bounds__(256) void fuse_kernel(
    const float* __restrict__ x, const float* __restrict__ scaleC,
    const float* __restrict__ shiftC, _Float16* __restrict__ img)
{
    __shared__ float rowv[CIN][HH];   // 33280 B
    __shared__ float T[CIN][24];      // 6144 B
    __shared__ float scs[CIN], shs[CIN];
    const int y = blockIdx.x % HH;
    const int b = blockIdx.x / HH;
    const int tx = threadIdx.x;

    if (tx < CIN) { scs[tx] = scaleC[tx]; shs[tx] = shiftC[tx]; }
    __syncthreads();

    for (int idx = tx; idx < CIN * HH; idx += 256) {
        const int c    = idx / HH;
        const int xcol = idx % HH;
        rowv[c][xcol] =
            scs[c] * x[((size_t)(b * CIN + c) * HH + y) * HH + xcol] + shs[c];
    }
    __syncthreads();

    for (int idx = tx; idx < CIN * 24; idx += 256) {
        const int c  = idx / 24;
        const int jg = idx % 24;
        const int j  = jg >> 3, g = jg & 7;
        const int o  = g * 16 + j;               // max 112+2, +15 -> 129 OK
        float s = 0.f;
        #pragma unroll
        for (int t = 0; t < 16; ++t) s += fabsf(rowv[c][o + t]);
        T[c][jg] = s * (1.f / 16.f);
    }
    __syncthreads();

    const float ni = (y == 0 || y == HH - 1) ? 1.f
                   : ((y == 1 || y == HH - 2) ? 2.f : 3.f);
    _Float16* orow = img + (size_t)(b * HH + y) * HH * CIN;
    for (int idx = tx; idx < CIN * HH; idx += 256) {
        const int xcol = idx / CIN;
        const int c    = idx % CIN;
        const float v  = rowv[c][xcol];
        const float sgn = (v > 0.f) ? 1.f : ((v < 0.f) ? -1.f : 0.f);
        float s = 0.f;
        #pragma unroll
        for (int j = 0; j < 3; ++j) {
            const int ox = xcol - j;
            if (ox >= 0 && ox <= 127) s += T[c][j * 8 + (ox >> 4)];
        }
        orow[idx] = (_Float16)(sgn * ni * s);    // contiguous, coalesced
    }
}

// ----- Kernel 4: 3x3 conv pad=1, LDS-free implicit GEMM with WMMA ----------
// Each wave: 16 co x 16 px x 4 rows. Fragments loaded directly from global.
union HF { v16h h; u32x4 q[2]; };

__global__ __launch_bounds__(256) void conv_wmma_kernel(
    const _Float16* __restrict__ img,   // [8][130][130][64] f16
    const _Float16* __restrict__ wh,    // [9][128][64] f16
    const float* __restrict__ bias,     // [128]
    float* __restrict__ out)            // [8][128][130][130] f32
{
    const int lane = threadIdx.x & 31;
    const int wave = threadIdx.x >> 5;
    const int m    = lane & 15;
    const int half = lane >> 4;

    int W = blockIdx.x * 8 + wave;      // 8*9*33*8 = 19008 waves
    const int coTile = W & 7;  W >>= 3;
    const int xt   = W % 9;    W /= 9;
    const int quad = W % 33;   W /= 33;
    const int b    = W;
    const int x0 = xt * 16;
    const int y0 = quad * 4;
    const int coBase = coTile * 16;

    // per-lane constant element offsets (f16 units)
    const int aLane = (coBase + m) * CIN + half * 8;   // + tap*COUT*CIN + cb (+16)
    const int pLane = (x0 + m) * CIN + half * 8;       // + rowbase*CIN + cb
    const bool xInImg = (x0 + m) < HH;

    const u32x4 zero = {0u, 0u, 0u, 0u};
    v8f acc[4];
    #pragma unroll
    for (int s = 0; s < 4; ++s) acc[s] = v8f{};

    for (int tap = 0; tap < 9; ++tap) {
        const int di = tap / 3 - 1;
        const int dj = tap % 3 - 1;
        const int xx   = x0 + m + dj;
        const bool xok = xInImg && (xx >= 0) && (xx < HH);
        #pragma unroll
        for (int cb = 0; cb < CIN; cb += 32) {
            HF a;
            const int aIdx = tap * (COUT * CIN) + aLane + cb;
            a.q[0] = *(const u32x4*)(wh + aIdx);
            a.q[1] = *(const u32x4*)(wh + aIdx + 16);
            #pragma unroll
            for (int s = 0; s < 4; ++s) {
                const int yy = y0 + s + di;          // wave-uniform
                if (yy >= 0 && yy < HH) {
                    HF bf;
                    bf.q[0] = zero; bf.q[1] = zero;
                    if (xok) {
                        const int bIdx =
                            ((b * HH + yy) * HH + dj) * CIN + cb + pLane;
                        bf.q[0] = *(const u32x4*)(img + bIdx);
                        bf.q[1] = *(const u32x4*)(img + bIdx + 16);
                    }
                    acc[s] = __builtin_amdgcn_wmma_f32_16x16x32_f16(
                        false, a.h, false, bf.h, (short)0, acc[s],
                        false, false);
                }
            }
        }
    }

    // Epilogue: D layout -> lane: n = lane%16, VGPR r: co = half*8 + r
    float bs[8];
    #pragma unroll
    for (int r = 0; r < 8; ++r) bs[r] = bias[coBase + half * 8 + r];

    const int xo = x0 + m;
    #pragma unroll
    for (int s = 0; s < 4; ++s) {
        const int y = y0 + s;
        if (y < HH && xo < HH) {
            #pragma unroll
            for (int r = 0; r < 8; ++r) {
                const int co = coBase + half * 8 + r;
                float v = acc[s][r] + bs[r];
                v = v > 0.f ? v : 0.f;
                out[((size_t)(b * COUT + co) * HH + y) * HH + xo] = v;
            }
        }
    }
}

// ---------------------------------------------------------------------------
extern "C" void kernel_launch(void* const* d_in, const int* in_sizes, int n_in,
                              void* d_out, int out_size, void* d_ws, size_t ws_size,
                              hipStream_t stream) {
    (void)in_sizes; (void)n_in; (void)out_size; (void)ws_size;
    const float* x      = (const float*)d_in[0];   // [8,64,130,130]
    const float* gamma  = (const float*)d_in[1];   // [64]
    const float* beta   = (const float*)d_in[2];   // [64]
    const float* conv_w = (const float*)d_in[3];   // [128,64,3,3]
    const float* conv_b = (const float*)d_in[4];   // [128]
    float* out = (float*)d_out;                    // [8,128,130,130]

    // Workspace layout
    char* ws = (char*)d_ws;
    float*    scaleC = (float*)ws;                          // 64 f32
    float*    shiftC = scaleC + 64;                         // 64 f32
    _Float16* wh     = (_Float16*)(ws + 512);               // 9*128*64 f16
    _Float16* img    = (_Float16*)(ws + 512 + (size_t)9 * COUT * CIN * 2);
    // img offset = 512 + 147456 = 147968 (16B aligned); img = 17.3 MB f16

    // 1) BN statistics -> per-channel scale/shift
    bnstats_kernel<<<CIN, 256, 0, stream>>>(x, gamma, beta, scaleC, shiftC);

    // 2) Weight conversion f32 -> f16, [tap][co][ci]
    const int wn = 9 * COUT * CIN;
    wprep_kernel<<<(wn + 255) / 256, 256, 0, stream>>>(conv_w, wh);

    // 3) Fused BN + im2col + bin_active + col2im -> img (f16 channel-last)
    fuse_kernel<<<8 * HH, 256, 0, stream>>>(x, scaleC, shiftC, img);

    // 4) 3x3 conv (pad=1) + bias + ReLU, LDS-free WMMA implicit GEMM
    conv_wmma_kernel<<<(8 * 9 * 33 * 8) / 8, 256, 0, stream>>>(img, wh, conv_b, out);
}
